// LSTMModel_30683246362735
// MI455X (gfx1250) — matrix operand order repaired and checked by reading, not compiled, not run on hardware
//
#include <hip/hip_runtime.h>
#include <hip/hip_fp16.h>

typedef _Float16 half_t;
typedef __attribute__((ext_vector_type(16))) _Float16 v16h;
typedef __attribute__((ext_vector_type(8)))  _Float16 v8h;
typedef __attribute__((ext_vector_type(8)))  float    v8f;

#define B_TOT   2048
#define T_STEPS 200
#define D_IN    4
#define H_DIM   256
#define O_DIM   6
#define MB      16          // batch tile per workgroup
#define KIN0    32          // x padded 4 -> 32
#define KA0     (KIN0 + H_DIM)   // 288, A row for layer 0 GEMM: [x_t | h1_{t-1}]
#define KA1     (H_DIM + H_DIM)  // 512, A row for layer 1 GEMM: [h1_t | h2_{t-1}]
#define KT0     (KA0 / 32)  // 9 k-tiles
#define KT1     (KA1 / 32)  // 16 k-tiles
#define NT      64          // 1024 gate outputs / 16

// ---------- fast activations (round-trip-safe at extremes) ----------
__device__ __forceinline__ float sigmoid_(float x) {
    return 1.0f / (1.0f + __expf(-x));
}
__device__ __forceinline__ float tanh_(float x) {
    x = fminf(fmaxf(x, -15.0f), 15.0f);
    float e = __expf(-2.0f * x);
    return (1.0f - e) / (1.0f + e);
}

// ---------- prep: pad+convert x [B,T,4]f32 -> [B,T,32]f16 ----------
__global__ void pack_x_kernel(const float* __restrict__ x, half_t* __restrict__ xp) {
    long idx = (long)blockIdx.x * blockDim.x + threadIdx.x;   // over B*T*32
    if (idx >= (long)B_TOT * T_STEPS * KIN0) return;
    int k  = (int)(idx & (KIN0 - 1));
    long bt = idx >> 5;
    float v = (k < D_IN) ? x[bt * D_IN + k] : 0.0f;
    xp[idx] = (half_t)v;
}

// ---------- prep: pack [W_ih ; W_hh] into WMMA B-fragment layout ----------
// Fragment = one 32x16 (KxN) f16 tile, stored as [32 lanes][16 halves] = 1 KB.
// Per-lane layout (self-consistent with main kernel): lane l -> n = l&15,
// khalf = l>>4; the 16 halves cover k = kt*32 + khalf*16 + 0..15 linearly.
// Fragment ORDER is wave-contiguous: linear frag f = (kt*8 + w)*8 + (g*2+p),
// where nt = g*16 + 2*w + p. So each wave reads one contiguous 8 KB chunk per
// k-tile using a single base address + immediate offsets.
__global__ void pack_w_kernel(const float* __restrict__ W_ih,
                              const float* __restrict__ W_hh,
                              int Din, int Kin, int Ktiles,
                              half_t* __restrict__ packed) {
    int t = blockIdx.x * blockDim.x + threadIdx.x;  // frag*32 + lane
    int nfrag = Ktiles * NT;
    if (t >= nfrag * 32) return;
    int lane = t & 31;
    int f    = t >> 5;            // linear fragment index
    int tile = f & 7;             // g*2 + p
    int wv   = (f >> 3) & 7;      // wave
    int kt   = f >> 6;            // k-tile
    int g    = tile >> 1;
    int p    = tile & 1;
    int nt   = g * 16 + 2 * wv + p;
    int n = lane & 15, gh = lane >> 4;
    int j = nt * 16 + n;                 // gate output column 0..1023
    half_t* dst = packed + (size_t)f * 512 + (size_t)lane * 16;
#pragma unroll
    for (int v = 0; v < 16; v++) {
        int k = kt * 32 + gh * 16 + v;
        float w;
        if (k < Kin) w = (k < Din) ? W_ih[(size_t)j * Din + k] : 0.0f;
        else         w = W_hh[(size_t)j * H_DIM + (k - Kin)];
        dst[v] = (half_t)w;
    }
}

// ---------- prep: combined bias ----------
__global__ void bias_kernel(const float* __restrict__ bi, const float* __restrict__ bh,
                            float* __restrict__ out) {
    int j = blockIdx.x * blockDim.x + threadIdx.x;
    if (j < 4 * H_DIM) out[j] = bi[j] + bh[j];
}

// ---------- A-fragment load, contiguous k order ----------
// lane 0-15: M=lane, K 0..15 contiguous; lanes 16-31: K 16..31 (matches packed B).
__device__ __forceinline__ v16h load_a(const half_t* __restrict__ A, int stride,
                                       int kt, int lane) {
    int m = lane & 15;
    int g = lane >> 4;
    const half_t* row = A + (size_t)m * stride + kt * 32;
    v8h lo = *(const v8h*)(row + g * 16);       // 16-byte aligned (stride mult of 8 halves)
    v8h hi = *(const v8h*)(row + g * 16 + 8);
    v16h a;
#pragma unroll
    for (int i = 0; i < 8; i++) { a[i] = lo[i]; a[i + 8] = hi[i]; }
    return a;
}

// ---------- fused 2-layer LSTM, persistent over all T ----------
__launch_bounds__(256)
__global__ void lstm2_fused_kernel(const half_t* __restrict__ xp,
                                   const half_t* __restrict__ pW0,
                                   const half_t* __restrict__ pW1,
                                   const float*  __restrict__ b0,
                                   const float*  __restrict__ b1,
                                   half_t* __restrict__ h2last) {
    __shared__ __align__(16) half_t A0[MB * KA0];  // [16][32 x | 256 h1]
    __shared__ __align__(16) half_t A1[MB * KA1];  // [16][256 h1 | 256 h2]

    const int lane  = threadIdx.x & 31;
    const int w     = threadIdx.x >> 5;     // wave 0..7
    const int n     = lane & 15;
    const int gHalf = lane >> 4;            // C layout: m = r + gHalf*8
    const int brow  = blockIdx.x * MB;

    // zero h regions of LDS (x region overwritten every step anyway)
    for (int i = threadIdx.x; i < MB * KA0; i += 256) A0[i] = (half_t)0.0f;
    for (int i = threadIdx.x; i < MB * KA1; i += 256) A1[i] = (half_t)0.0f;

    float c1[2][8], c2[2][8];
#pragma unroll
    for (int p = 0; p < 2; p++)
#pragma unroll
        for (int r = 0; r < 8; r++) { c1[p][r] = 0.0f; c2[p][r] = 0.0f; }

    // per-lane biases for this wave's 8 gate tiles (4 gates x 2 hidden tiles)
    float bia0[4][2], bia1[4][2];
#pragma unroll
    for (int g = 0; g < 4; g++)
#pragma unroll
        for (int p = 0; p < 2; p++) {
            int col = (g * 16 + (2 * w + p)) * 16 + n;
            bia0[g][p] = b0[col];
            bia1[g][p] = b1[col];
        }

    // per-wave, per-lane weight base pointers (chunk stride per kt = 64 frags * 512 halves)
    const half_t* w0base = pW0 + ((size_t)w * 8) * 512 + (size_t)lane * 16;
    const half_t* w1base = pW1 + ((size_t)w * 8) * 512 + (size_t)lane * 16;

    const v8f vzero = {};

    for (int t = 0; t < T_STEPS; t++) {
        // ---- stage x_t tile into A0 input slice (16 rows x 32 halves) ----
        {
            int row  = threadIdx.x >> 4;   // 16 dwords per row
            int colD = threadIdx.x & 15;
            const unsigned* src =
                (const unsigned*)(xp + ((size_t)(brow + row) * T_STEPS + t) * KIN0);
            unsigned* dst = (unsigned*)(A0 + (size_t)row * KA0);
            dst[colD] = src[colD];
            if (t + 1 < T_STEPS) {  // global_prefetch_b8 for next step's tile
                __builtin_prefetch(
                    (const char*)(xp + ((size_t)(brow + row) * T_STEPS + t + 1) * KIN0), 0, 3);
            }
        }
        __syncthreads();   // x_t visible; prior-step h2 writes/readers fenced by order

        // ---- GEMM 1: gates1[16,1024] = [x_t | h1]{16,288} x W0{288,1024} ----
        v8f acc[4][2];
#pragma unroll
        for (int g = 0; g < 4; g++)
#pragma unroll
            for (int p = 0; p < 2; p++) acc[g][p] = vzero;

#pragma unroll 1   // keep rolled: loads depend on kt -> cannot be hoisted/spilled
        for (int kt = 0; kt < KT0; kt++) {
            v16h afrag = load_a(A0, KA0, kt, lane);
            const half_t* bb = w0base + (size_t)kt * (NT * 512);  // this kt's chunk
#pragma unroll
            for (int g = 0; g < 4; g++)
#pragma unroll
                for (int p = 0; p < 2; p++) {
                    v16h bfrag = *(const v16h*)(bb + (g * 2 + p) * 512);
                    acc[g][p] = __builtin_amdgcn_wmma_f32_16x16x32_f16(
                        false, afrag, false, bfrag, (short)0, acc[g][p], false, false);
                }
        }
        __syncthreads();   // all waves done reading A0/A1 before h1 is written

        // ---- LSTM cell layer 0 -> h1 into A0 (next step) and A1 (layer 1 input) ----
#pragma unroll
        for (int p = 0; p < 2; p++) {
            int hcol = (2 * w + p) * 16 + n;
#pragma unroll
            for (int r = 0; r < 8; r++) {
                int mm  = r + gHalf * 8;
                float i_ = sigmoid_(acc[0][p][r] + bia0[0][p]);
                float f_ = sigmoid_(acc[1][p][r] + bia0[1][p]);
                float g_ = tanh_  (acc[2][p][r] + bia0[2][p]);
                float o_ = sigmoid_(acc[3][p][r] + bia0[3][p]);
                float c  = f_ * c1[p][r] + i_ * g_;
                c1[p][r] = c;
                half_t h = (half_t)(o_ * tanh_(c));
                A0[(size_t)mm * KA0 + KIN0 + hcol] = h;
                A1[(size_t)mm * KA1 + hcol]        = h;
            }
        }
        __syncthreads();   // h1 visible before GEMM 2 reads it

        // ---- GEMM 2: gates2[16,1024] = [h1_t | h2]{16,512} x W1{512,1024} ----
#pragma unroll
        for (int g = 0; g < 4; g++)
#pragma unroll
            for (int p = 0; p < 2; p++) acc[g][p] = vzero;

#pragma unroll 1
        for (int kt = 0; kt < KT1; kt++) {
            v16h afrag = load_a(A1, KA1, kt, lane);
            const half_t* bb = w1base + (size_t)kt * (NT * 512);
#pragma unroll
            for (int g = 0; g < 4; g++)
#pragma unroll
                for (int p = 0; p < 2; p++) {
                    v16h bfrag = *(const v16h*)(bb + (g * 2 + p) * 512);
                    acc[g][p] = __builtin_amdgcn_wmma_f32_16x16x32_f16(
                        false, afrag, false, bfrag, (short)0, acc[g][p], false, false);
                }
        }
        __syncthreads();   // all waves done reading A1's h2 slice before overwrite

        // ---- LSTM cell layer 1 -> h2 into A1 ----
#pragma unroll
        for (int p = 0; p < 2; p++) {
            int hcol = (2 * w + p) * 16 + n;
#pragma unroll
            for (int r = 0; r < 8; r++) {
                int mm  = r + gHalf * 8;
                float i_ = sigmoid_(acc[0][p][r] + bia1[0][p]);
                float f_ = sigmoid_(acc[1][p][r] + bia1[1][p]);
                float g_ = tanh_  (acc[2][p][r] + bia1[2][p]);
                float o_ = sigmoid_(acc[3][p][r] + bia1[3][p]);
                float c  = f_ * c2[p][r] + i_ * g_;
                c2[p][r] = c;
                half_t h = (half_t)(o_ * tanh_(c));
                A1[(size_t)mm * KA1 + H_DIM + hcol] = h;
                if (t == T_STEPS - 1)
                    h2last[(size_t)(brow + mm) * H_DIM + hcol] = h;
            }
        }
        // next iteration's first barrier fences these writes vs next GEMM 2
    }
}

// ---------- final FC: out[b,o] = h2_last[b,:] . W_fc[o,:] + b_fc[o] ----------
__global__ void fc_kernel(const half_t* __restrict__ h2last,
                          const float* __restrict__ W_fc,
                          const float* __restrict__ b_fc,
                          float* __restrict__ out) {
    int idx = blockIdx.x * blockDim.x + threadIdx.x;
    if (idx >= B_TOT * O_DIM) return;
    int b = idx / O_DIM, o = idx % O_DIM;
    const half_t* hrow = h2last + (size_t)b * H_DIM;
    const float*  wrow = W_fc   + (size_t)o * H_DIM;
    float s = b_fc[o];
#pragma unroll 8
    for (int j = 0; j < H_DIM; j++) s += (float)hrow[j] * wrow[j];
    out[idx] = s;
}

extern "C" void kernel_launch(void* const* d_in, const int* in_sizes, int n_in,
                              void* d_out, int out_size, void* d_ws, size_t ws_size,
                              hipStream_t stream) {
    (void)in_sizes; (void)n_in; (void)out_size; (void)ws_size;
    const float* x     = (const float*)d_in[0];
    const float* W_ih0 = (const float*)d_in[1];
    const float* W_hh0 = (const float*)d_in[2];
    const float* b_ih0 = (const float*)d_in[3];
    const float* b_hh0 = (const float*)d_in[4];
    const float* W_ih1 = (const float*)d_in[5];
    const float* W_hh1 = (const float*)d_in[6];
    const float* b_ih1 = (const float*)d_in[7];
    const float* b_hh1 = (const float*)d_in[8];
    const float* W_fc  = (const float*)d_in[9];
    const float* b_fc  = (const float*)d_in[10];
    float* out = (float*)d_out;

    // workspace carve-up (256B aligned)
    char* ws = (char*)d_ws;
    size_t off = 0;
    auto carve = [&](size_t bytes) -> char* {
        char* p = ws + off;
        off = (off + bytes + 255) & ~(size_t)255;
        return p;
    };
    half_t* pW0    = (half_t*)carve((size_t)KT0 * NT * 1024);            // 576 KB
    half_t* pW1    = (half_t*)carve((size_t)KT1 * NT * 1024);            // 1 MB
    float*  b0     = (float*) carve(4 * H_DIM * sizeof(float));
    float*  b1     = (float*) carve(4 * H_DIM * sizeof(float));
    half_t* xp     = (half_t*)carve((size_t)B_TOT * T_STEPS * KIN0 * 2); // 25 MB
    half_t* h2last = (half_t*)carve((size_t)B_TOT * H_DIM * 2);          // 1 MB

    // prep
    {
        long nx = (long)B_TOT * T_STEPS * KIN0;
        pack_x_kernel<<<(unsigned)((nx + 255) / 256), 256, 0, stream>>>(x, xp);
    }
    pack_w_kernel<<<(KT0 * NT * 32 + 255) / 256, 256, 0, stream>>>(
        W_ih0, W_hh0, D_IN, KIN0, KT0, pW0);
    pack_w_kernel<<<(KT1 * NT * 32 + 255) / 256, 256, 0, stream>>>(
        W_ih1, W_hh1, H_DIM, H_DIM, KT1, pW1);
    bias_kernel<<<(4 * H_DIM + 255) / 256, 256, 0, stream>>>(b_ih0, b_hh0, b0);
    bias_kernel<<<(4 * H_DIM + 255) / 256, 256, 0, stream>>>(b_ih1, b_hh1, b1);

    // fused recurrent core: 128 persistent blocks, 8 waves each
    lstm2_fused_kernel<<<B_TOT / MB, 256, 0, stream>>>(xp, pW0, pW1, b0, b1, h2last);

    // tiny head
    fc_kernel<<<(B_TOT * O_DIM + 255) / 256, 256, 0, stream>>>(h2last, W_fc, b_fc, out);
}
